// MoERouter_24283745091734
// MI455X (gfx1250) — matrix operand order repaired and checked by reading, not compiled, not run on hardware
//
#include <hip/hip_runtime.h>

typedef __attribute__((ext_vector_type(16))) __bf16  v16bf;
typedef __attribute__((ext_vector_type(4)))  __bf16  v4bf;
typedef __attribute__((ext_vector_type(8)))  float   v8f;
typedef __attribute__((ext_vector_type(4)))  float   f4;

static constexpr int S_TOK = 16384;
static constexpr int D_DIM = 2048;
static constexpr int E_EXP = 256;
static constexpr int K_TOP = 8;
static constexpr int TKG   = 4;

__device__ __forceinline__ float sigmoidf_fast(float v) {
  // router weights are renormalized afterwards; hardware rcp is plenty
  return __builtin_amdgcn_rcpf(1.0f + __expf(-v));
}

// -----------------------------------------------------------------------------
// Kernel 0: one-time fp32 -> bf16 conversion of the 2 MB router weight.
// -----------------------------------------------------------------------------
__global__ __launch_bounds__(256)
void convert_w_bf16(const float* __restrict__ W, __bf16* __restrict__ Wbf)
{
  const int i = (blockIdx.x * 256 + threadIdx.x) * 4;
  f4 v = *(const f4*)(W + i);
  v4bf o;
  o[0] = (__bf16)v.x; o[1] = (__bf16)v.y; o[2] = (__bf16)v.z; o[3] = (__bf16)v.w;
  *(v4bf*)(Wbf + i) = o;
}

// Build one 16x32 bf16 A fragment from fp32 x (contiguous K layout, matches B):
// lane half h: elems 0..7  <- K = k0 + 16h .. k0+16h+7
//              elems 8..15 <- K = k0+16h+8 .. k0+16h+15
__device__ __forceinline__ v16bf load_a_frag(const float* __restrict__ xrow,
                                             int k0, int half)
{
  const f4* pa0 = (const f4*)(xrow + k0 + half * 16);
  const f4* pa1 = (const f4*)(xrow + k0 + 8 + half * 16);
  f4 a00 = pa0[0], a01 = pa0[1];
  f4 a10 = pa1[0], a11 = pa1[1];
  v16bf A;
  A[0]=(__bf16)a00.x;  A[1]=(__bf16)a00.y;  A[2]=(__bf16)a00.z;  A[3]=(__bf16)a00.w;
  A[4]=(__bf16)a01.x;  A[5]=(__bf16)a01.y;  A[6]=(__bf16)a01.z;  A[7]=(__bf16)a01.w;
  A[8]=(__bf16)a10.x;  A[9]=(__bf16)a10.y;  A[10]=(__bf16)a10.z; A[11]=(__bf16)a10.w;
  A[12]=(__bf16)a11.x; A[13]=(__bf16)a11.y; A[14]=(__bf16)a11.z; A[15]=(__bf16)a11.w;
  return A;
}

// -----------------------------------------------------------------------------
// Kernel 1: scores[s, e] = sigmoid( sum_d x[s,d] * W[e,d] )
// Block = 256 threads (8 waves) covering 32 token rows x all 256 experts.
// Wave layout: (wave>>2) selects 16-row stripe, (wave&3) selects a 64-col quad
// held as four 16x16 accumulators -> 4 WMMAs per A fragment per k-step.
// Software-pipelined: iteration k issues loads/converts for k+1 before the
// WMMAs consume the current fragments, so the per-iteration s_wait hides
// behind 4 WMMAs instead of stalling each one.
// -----------------------------------------------------------------------------
__global__ __launch_bounds__(256, 2)
void router_gemm_sigmoid(const float* __restrict__ x,
                         const __bf16* __restrict__ Wbf,
                         float* __restrict__ scores)
{
  const int lane = threadIdx.x & 31;
  const int wave = threadIdx.x >> 5;
  const int half = lane >> 4;     // 0: lanes 0-15, 1: lanes 16-31
  const int l16  = lane & 15;
  const int rowHalf = wave >> 2;
  const int colQuad = wave & 3;
  const int rowBase = blockIdx.x * 32 + rowHalf * 16;
  const int colBase = colQuad * 64;

  const float*  xrow = x + (size_t)(rowBase + l16) * D_DIM;
  const __bf16* wb0  = Wbf + (size_t)(colBase +  0 + l16) * D_DIM;
  const __bf16* wb1  = Wbf + (size_t)(colBase + 16 + l16) * D_DIM;
  const __bf16* wb2  = Wbf + (size_t)(colBase + 32 + l16) * D_DIM;
  const __bf16* wb3  = Wbf + (size_t)(colBase + 48 + l16) * D_DIM;

  v8f c0 = {}, c1 = {}, c2 = {}, c3 = {};

  const int kboff = half * 16;   // B: lane half h holds K = k0+16h .. k0+16h+15

  // ---- pipeline prologue: fragments for k0 = 0 ----
  v16bf A  = load_a_frag(xrow, 0, half);
  v16bf B0 = *(const v16bf*)(wb0 + kboff);
  v16bf B1 = *(const v16bf*)(wb1 + kboff);
  v16bf B2 = *(const v16bf*)(wb2 + kboff);
  v16bf B3 = *(const v16bf*)(wb3 + kboff);

  #pragma unroll 4
  for (int k0 = 0; k0 < D_DIM - 32; k0 += 32) {
    const int kn = k0 + 32;
    // issue next-iteration loads (+ conversions) first
    v16bf An  = load_a_frag(xrow, kn, half);
    v16bf B0n = *(const v16bf*)(wb0 + kn + kboff);
    v16bf B1n = *(const v16bf*)(wb1 + kn + kboff);
    v16bf B2n = *(const v16bf*)(wb2 + kn + kboff);
    v16bf B3n = *(const v16bf*)(wb3 + kn + kboff);
    __builtin_prefetch(xrow + k0 + 512, 0, 3);   // x stream, 2 KB ahead

    // consume current fragments (independent of the loads above)
    c0 = __builtin_amdgcn_wmma_f32_16x16x32_bf16(false, A, false, B0,
                                                 (short)0, c0, false, false);
    c1 = __builtin_amdgcn_wmma_f32_16x16x32_bf16(false, A, false, B1,
                                                 (short)0, c1, false, false);
    c2 = __builtin_amdgcn_wmma_f32_16x16x32_bf16(false, A, false, B2,
                                                 (short)0, c2, false, false);
    c3 = __builtin_amdgcn_wmma_f32_16x16x32_bf16(false, A, false, B3,
                                                 (short)0, c3, false, false);

    A = An; B0 = B0n; B1 = B1n; B2 = B2n; B3 = B3n;
  }

  // ---- pipeline epilogue: last k-step ----
  c0 = __builtin_amdgcn_wmma_f32_16x16x32_bf16(false, A, false, B0,
                                               (short)0, c0, false, false);
  c1 = __builtin_amdgcn_wmma_f32_16x16x32_bf16(false, A, false, B1,
                                               (short)0, c1, false, false);
  c2 = __builtin_amdgcn_wmma_f32_16x16x32_bf16(false, A, false, B2,
                                               (short)0, c2, false, false);
  c3 = __builtin_amdgcn_wmma_f32_16x16x32_bf16(false, A, false, B3,
                                               (short)0, c3, false, false);

  // C/D layout: element r of v8f -> M = rowBase + 8*half + r, N = l16
  const int mBase = rowBase + half * 8;
  #pragma unroll
  for (int r = 0; r < 8; ++r) {
    float* srow = scores + (size_t)(mBase + r) * E_EXP + colBase + l16;
    srow[0]  = sigmoidf_fast(c0[r]);
    srow[16] = sigmoidf_fast(c1[r]);
    srow[32] = sigmoidf_fast(c2[r]);
    srow[48] = sigmoidf_fast(c3[r]);
  }
}

// -----------------------------------------------------------------------------
// Kernel 2: group-limited greedy top-k routing. One wave32 per token.
// Lane t owns experts [8t, 8t+8); each group of 32 experts = lane quad (t>>2).
// -----------------------------------------------------------------------------
__global__ __launch_bounds__(256, 4)
void router_topk(const float* __restrict__ scores,
                 const float* __restrict__ e_bias,
                 float* __restrict__ out_idx,
                 float* __restrict__ out_w)
{
  const int lane = threadIdx.x & 31;
  const int wave = threadIdx.x >> 5;
  const int s = blockIdx.x * 8 + wave;

  float raw[8], bm[8];
  const float* sp = scores + (size_t)s * E_EXP + lane * 8;
  #pragma unroll
  for (int j = 0; j < 8; ++j) {
    raw[j] = sp[j];
    bm[j]  = raw[j] + e_bias[lane * 8 + j];   // biased scores (routing only)
  }

  // ---- per-group score: sum of top-2 biased scores within group ----
  float m1 = -INFINITY, m2 = -INFINITY;
  #pragma unroll
  for (int j = 0; j < 8; ++j) {
    float v = bm[j];
    if (v > m1) { m2 = m1; m1 = v; }
    else if (v > m2) { m2 = v; }
  }
  // merge top-2 pairs across the 4 lanes of this group (xor stays in quad)
  #pragma unroll
  for (int m = 1; m <= 2; m <<= 1) {
    float o1 = __shfl_xor(m1, m);
    float o2 = __shfl_xor(m2, m);
    float n1 = fmaxf(m1, o1);
    float n2 = fmaxf(fminf(m1, o1), fmaxf(m2, o2));
    m1 = n1; m2 = n2;
  }
  const float gs  = m1 + m2;     // group score, identical in all 4 quad lanes
  const int   myg = lane >> 2;

  // ---- rank my group among 8 (ties -> lower group index, like lax.top_k) ----
  int rank = 0;
  #pragma unroll
  for (int g0 = 0; g0 < 8; ++g0) {
    float o = __shfl(gs, g0 * 4);
    if (o > gs || (o == gs && g0 < myg)) rank++;
  }
  if (rank >= TKG) {
    #pragma unroll
    for (int j = 0; j < 8; ++j) bm[j] = -INFINITY;   // mask non-selected groups
  }

  // ---- top-8 experts among the 4 selected groups ----
  float sum = 0.0f;
  int   myI = 0;
  float myW = 0.0f;
  #pragma unroll
  for (int k = 0; k < K_TOP; ++k) {
    float lv = -INFINITY;
    int   li = 0x7FFFFFFF;
    #pragma unroll
    for (int j = 0; j < 8; ++j) {
      if (bm[j] > lv) { lv = bm[j]; li = lane * 8 + j; }
    }
    // wave-wide argmax, ties -> lower expert index
    #pragma unroll
    for (int m = 16; m >= 1; m >>= 1) {
      float ov = __shfl_xor(lv, m);
      int   oi = __shfl_xor(li, m);
      if (ov > lv || (ov == lv && oi < li)) { lv = ov; li = oi; }
    }
    const int ow = li >> 3, oj = li & 7;
    if (ow == lane) {                       // owner removes the winner
      #pragma unroll
      for (int j = 0; j < 8; ++j) if (j == oj) bm[j] = -INFINITY;
    }
    // gather UNbiased score of the winner (register select + broadcast)
    float rv = raw[0];
    #pragma unroll
    for (int j = 1; j < 8; ++j) if (j == oj) rv = raw[j];
    const float rw = __shfl(rv, ow);
    sum += rw;
    if (lane == k) { myI = li; myW = rw; }
  }

  if (lane < K_TOP) {
    out_idx[(size_t)s * K_TOP + lane] = (float)myI;
    out_w  [(size_t)s * K_TOP + lane] = myW / (sum + 1e-20f);
  }
}

// -----------------------------------------------------------------------------
// d_out layout (reference return order, flat): [S*K idx][S*K weights][S*E scores]
// d_ws: bf16 copy of W (E*D*2 = 1 MB)
// -----------------------------------------------------------------------------
extern "C" void kernel_launch(void* const* d_in, const int* in_sizes, int n_in,
                              void* d_out, int out_size, void* d_ws, size_t ws_size,
                              hipStream_t stream) {
  (void)in_sizes; (void)n_in; (void)out_size; (void)ws_size;

  const float* x      = (const float*)d_in[0];
  const float* W      = (const float*)d_in[1];
  const float* e_bias = (const float*)d_in[2];

  float* out     = (float*)d_out;
  float* out_idx = out;
  float* out_w   = out + (size_t)S_TOK * K_TOP;
  float* sc      = out + (size_t)2 * S_TOK * K_TOP;   // scores live in d_out

  __bf16* Wbf = (__bf16*)d_ws;

  convert_w_bf16<<<(E_EXP * D_DIM) / (256 * 4), 256, 0, stream>>>(W, Wbf);
  router_gemm_sigmoid<<<S_TOK / 32, 256, 0, stream>>>(x, Wbf, sc);
  router_topk<<<S_TOK / 8, 256, 0, stream>>>(sc, e_bias, out_idx, out_w);
}
